// GRACE_20736102105704
// MI455X (gfx1250) — matrix operand order repaired and checked by reading, compiled, not run on hardware
//
#include <hip/hip_runtime.h>

// ---------------------------------------------------------------------------
// SimCLR NT-Xent loss, fused for MI455X (gfx1250, wave32, WMMA).
//   n = normalize(proj([z1;z2]))           [16384 x 256]  f16
//   S_i = sum_j exp(dot(n_i,n_j)/tau)      (exp+rowsum fused into WMMA epilogue)
//   out_i = 0.5*(log(S_i - e^2) + log(S_{N+i} - e^2)) - dot(n1_i,n2_i)/tau
// sim_kernel: 2 M-tiles/wave (B-fragment reused by 2 WMMAs), 32-wide panels,
// double-buffered LDS fed by ASYNC global->LDS DMA (no VGPR staging),
// column-split grid + f32 global atomics.
// ---------------------------------------------------------------------------

typedef __attribute__((ext_vector_type(16))) _Float16 v16h;
typedef __attribute__((ext_vector_type(8)))  _Float16 v8h;
typedef __attribute__((ext_vector_type(8)))  float    v8f;
typedef __attribute__((__vector_size__(4 * sizeof(int)))) int v4i;

#define AS1 __attribute__((address_space(1)))
#define AS3 __attribute__((address_space(3)))

#define NROW 8192
#define DIM  256
#define M2   16384                       // 2*NROW
#define SIM_SCALE 2.8853900817779268f    // log2(e)/tau, tau = 0.5
#define INV_TAU   2.0f
#define E2        7.389056098930650f     // exp(1/tau)

#define PANEL_W   32                     // columns per LDS panel
#define COL_HALF  (M2 / 2)               // column split per block
#define NPAN      (COL_HALF / PANEL_W)   // 256 panels per block

#if __has_builtin(__builtin_amdgcn_global_load_async_to_lds_b128)
#define USE_ASYNC_LDS 1
#else
#define USE_ASYNC_LDS 0
#endif

// ---- A fragment: 16x32 f16 tile, rows M0..M0+15 (base pre-offset), K k0..k0+31.
// ISA layout: lanes 0-15 hold K {0..7,16..23}; lanes 16-31 hold K {8..15,24..31}.
__device__ __forceinline__ v16h load_a_frag(const _Float16* __restrict__ base,
                                            int ld, int k0) {
  int lane = threadIdx.x & 31;
  int row  = lane & 15;
  int kb   = k0 + ((lane >> 4) << 3);
  const _Float16* p = base + (size_t)row * ld + kb;
  v8h lo = *(const v8h*)(p);
  v8h hi = *(const v8h*)(p + 16);
  v16h a;
#pragma unroll
  for (int i = 0; i < 8; ++i) { a[i] = lo[i]; a[8 + i] = hi[i]; }
  return a;
}

// ---- B fragment from a K-major packed matrix wp[k*ldn + n]:
// lane L supplies column N0+L%16; lanes 0-15 hold K {0..7,16..23}, lanes 16-31 K {8..15,24..31}.
__device__ __forceinline__ v16h load_b_frag_g(const _Float16* __restrict__ wp,
                                              int ldn, int k0, int N0) {
  int lane = threadIdx.x & 31;
  int n = N0 + (lane & 15);
  int kb = k0 + ((lane >> 4) << 3);
  v16h b;
#pragma unroll
  for (int i = 0; i < 8; ++i) {
    b[i] = wp[(size_t)(kb + i) * ldn + n];
    b[8 + i] = wp[(size_t)(kb + 16 + i) * ldn + n];
  }
  return b;
}

// ---- B fragment from LDS panel laid out panel[k*PANEL_W + n], k in [0,256).
__device__ __forceinline__ v16h load_b_frag_lds(const _Float16* panel, int k0, int nt) {
  int lane = threadIdx.x & 31;
  int n = nt + (lane & 15);
  int kb = k0 + ((lane >> 4) << 3);
  v16h b;
#pragma unroll
  for (int i = 0; i < 8; ++i) {
    b[i] = panel[(kb + i) * PANEL_W + n];
    b[8 + i] = panel[(kb + 16 + i) * PANEL_W + n];
  }
  return b;
}

__device__ __forceinline__ v8f wmma_f16(v16h a, v16h b, v8f c) {
  return __builtin_amdgcn_wmma_f32_16x16x32_f16(false, a, false, b,
                                                (short)0, c, false, false);
}

#if USE_ASYNC_LDS
// Async DMA: global (AS1) -> LDS (AS3), 16B, tracked by ASYNCcnt.
// Param types per compiler diagnostic: (int4 __device__*, int4 __shared__*, imm, imm).
// Generic->AS casts go through integers: LDS generic VAs truncate to the
// 32-bit LDS offset (ISA aperture rules); global generic VAs are the 64b VA.
__device__ __forceinline__ void async_copy_b128(const _Float16* g, _Float16* l) {
  AS1 v4i* gp = (AS1 v4i*)(unsigned long long)(__UINTPTR_TYPE__)g;
  AS3 v4i* lp = (AS3 v4i*)(unsigned int)(__UINTPTR_TYPE__)l;
  __builtin_amdgcn_global_load_async_to_lds_b128(gp, lp, 0, 0);
}
__device__ __forceinline__ void wait_asynccnt0() {
#if __has_builtin(__builtin_amdgcn_s_wait_asynccnt)
  __builtin_amdgcn_s_wait_asynccnt(0);
#else
  asm volatile("s_wait_asynccnt 0" ::: "memory");
#endif
}
#endif

// ===========================================================================
// Kernel 0: fp32 -> f16 conversion + weight transposition + S zero-init.
// ===========================================================================
__global__ void prep_kernel(const float* __restrict__ z1, const float* __restrict__ z2,
                            const float* __restrict__ W1, const float* __restrict__ W2,
                            _Float16* __restrict__ zh,
                            _Float16* __restrict__ w1p, _Float16* __restrict__ w2p,
                            float* __restrict__ S) {
  size_t i = (size_t)blockIdx.x * blockDim.x + threadIdx.x;
  const size_t NZ = (size_t)NROW * DIM;
  if (i < NZ)           zh[i] = (_Float16)z1[i];
  else if (i < 2 * NZ)  zh[i] = (_Float16)z2[i - NZ];
  if (i < (size_t)DIM * DIM) {
    int k = (int)(i >> 8), n = (int)(i & 255);
    w1p[k * DIM + n] = (_Float16)W1[n * DIM + k];  // W1 [P][D] -> [D][P]
    w2p[k * DIM + n] = (_Float16)W2[n * DIM + k];  // W2 [D][P] -> [P][D]
  }
  if (i < (size_t)M2) S[i] = 0.0f;                 // sim_kernel accumulates atomically
}

// ===========================================================================
// Kernel 1: H = elu(Z @ W1^T + b1).  8 waves/block, 16 rows/wave.
// ===========================================================================
__global__ void __launch_bounds__(256)
proj1_kernel(const _Float16* __restrict__ zh, const _Float16* __restrict__ w1p,
             const float* __restrict__ b1, _Float16* __restrict__ hh) {
  int wave = threadIdx.x >> 5, lane = threadIdx.x & 31;
  int M0 = blockIdx.x * 128 + wave * 16;
  v16h A[8];
#pragma unroll
  for (int kc = 0; kc < 8; ++kc) A[kc] = load_a_frag(zh + (size_t)M0 * DIM, DIM, kc * 32);
  int nsub = lane & 15, mhalf = (lane >> 4) << 3;
  for (int j = 0; j < 16; ++j) {
    int N0 = j * 16;
    v8f acc = {};
#pragma unroll
    for (int kc = 0; kc < 8; ++kc)
      acc = wmma_f16(A[kc], load_b_frag_g(w1p, DIM, kc * 32, N0), acc);
    float bv = b1[N0 + nsub];
#pragma unroll
    for (int r = 0; r < 8; ++r) {
      float x = acc[r] + bv;
      x = (x > 0.0f) ? x : (__expf(x) - 1.0f);           // ELU(alpha=1)
      hh[(size_t)(M0 + r + mhalf) * DIM + N0 + nsub] = (_Float16)x;
    }
  }
}

// ===========================================================================
// Kernel 2: P = H @ W2^T + b2; row-normalize; emit nh (row-major) and nT
// (K-major transpose). 4 waves/block, strip buffered in 64KB LDS.
// ===========================================================================
__global__ void __launch_bounds__(128)
proj2_kernel(const _Float16* __restrict__ hh, const _Float16* __restrict__ w2p,
             const float* __restrict__ b2,
             _Float16* __restrict__ nh, _Float16* __restrict__ nT) {
  __shared__ float strip[4 * 16 * DIM];   // 64 KB
  __shared__ float invn[4 * 16];
  int wave = threadIdx.x >> 5, lane = threadIdx.x & 31;
  int M0 = blockIdx.x * 64 + wave * 16;
  float* ws = strip + wave * 16 * DIM;
  v16h A[8];
#pragma unroll
  for (int kc = 0; kc < 8; ++kc) A[kc] = load_a_frag(hh + (size_t)M0 * DIM, DIM, kc * 32);
  int nsub = lane & 15, mhalf = (lane >> 4) << 3;
  float ssq[8];
#pragma unroll
  for (int r = 0; r < 8; ++r) ssq[r] = 0.0f;
  for (int j = 0; j < 16; ++j) {
    int N0 = j * 16;
    v8f acc = {};
#pragma unroll
    for (int kc = 0; kc < 8; ++kc)
      acc = wmma_f16(A[kc], load_b_frag_g(w2p, DIM, kc * 32, N0), acc);
    float bv = b2[N0 + nsub];
#pragma unroll
    for (int r = 0; r < 8; ++r) {
      float x = acc[r] + bv;
      ws[(r + mhalf) * DIM + N0 + nsub] = x;
      ssq[r] += x * x;
    }
  }
#pragma unroll
  for (int r = 0; r < 8; ++r)
    for (int off = 1; off < 16; off <<= 1) ssq[r] += __shfl_xor(ssq[r], off, 32);
  if (nsub == 0) {
#pragma unroll
    for (int r = 0; r < 8; ++r)
      invn[wave * 16 + r + mhalf] = 1.0f / fmaxf(__fsqrt_rn(ssq[r]), 1e-12f);
  }
  __syncthreads();
  for (int r16 = 0; r16 < 16; ++r16) {
    float iv = invn[wave * 16 + r16];
    int m = M0 + r16;
    for (int c = lane; c < DIM; c += 32) {
      float v = ws[r16 * DIM + c] * iv;
      nh[(size_t)m * DIM + c] = (_Float16)v;
      nT[(size_t)c * M2 + m]  = (_Float16)v;
    }
  }
}

// ===========================================================================
// Kernel 3: fused exp-rowsum similarity GEMM.
// 4 waves x 32 rows = 128 rows/block; grid split 2x over columns (256 blocks).
// Per panel: 32 WMMAs/wave, each B fragment reused by 2 M-tiles.
// Panel p+1 DMAs into the other LDS buffer while panel p computes.
// ===========================================================================
__global__ void __launch_bounds__(128)
sim_kernel(const _Float16* __restrict__ nh, const _Float16* __restrict__ nT,
           float* __restrict__ S) {
#if USE_ASYNC_LDS
  __shared__ _Float16 panel[2][DIM * PANEL_W];  // 2 x 16 KB, double-buffered
#else
  __shared__ _Float16 panel[1][DIM * PANEL_W];  // 16 KB, reg-staged pipeline
#endif
  int wave = threadIdx.x >> 5, lane = threadIdx.x & 31;
  int t = threadIdx.x;
  int C0 = (blockIdx.x & 1) * COL_HALF;
  int M0 = (blockIdx.x >> 1) * 128 + wave * 32;

  v16h A[2][8];                               // 2 M-tiles resident (128 VGPRs)
#pragma unroll
  for (int mt = 0; mt < 2; ++mt)
#pragma unroll
    for (int kc = 0; kc < 8; ++kc)
      A[mt][kc] = load_a_frag(nh + (size_t)(M0 + mt * 16) * DIM, DIM, kc * 32);

  float sums[2][8];
#pragma unroll
  for (int mt = 0; mt < 2; ++mt)
#pragma unroll
    for (int r = 0; r < 8; ++r) sums[mt][r] = 0.0f;

  // 16KB panel = 1024 x 16B chunks, 8 per thread; chunk c = j*128 + t
#define CHUNK_K(c)   ((c) >> 2)
#define CHUNK_OFF(c) (((c) & 3) * 8)

#if USE_ASYNC_LDS
#define ISSUE_PANEL(buf, N0)                                              \
  {                                                                       \
    _Pragma("unroll")                                                     \
    for (int j = 0; j < 8; ++j) {                                         \
      int c = j * 128 + t;                                                \
      async_copy_b128(nT + (size_t)CHUNK_K(c) * M2 + (N0) + CHUNK_OFF(c), \
                      &panel[buf][CHUNK_K(c) * PANEL_W + CHUNK_OFF(c)]);  \
    }                                                                     \
  }
  ISSUE_PANEL(0, C0);
  wait_asynccnt0();
  __syncthreads();
#else
  v8h stg[8];                                 // staging regs (32 VGPRs)
#define LOAD_STAGE(N0)                                                    \
  {                                                                       \
    _Pragma("unroll")                                                     \
    for (int j = 0; j < 8; ++j) {                                         \
      int c = j * 128 + t;                                                \
      stg[j] = *(const v8h*)(nT + (size_t)CHUNK_K(c) * M2 + (N0) + CHUNK_OFF(c)); \
    }                                                                     \
  }
  LOAD_STAGE(C0);
#endif

  for (int p = 0; p < NPAN; ++p) {
#if USE_ASYNC_LDS
    const _Float16* cur = panel[p & 1];
    if (p + 1 < NPAN) ISSUE_PANEL((p + 1) & 1, C0 + (p + 1) * PANEL_W);
#else
    __syncthreads();                          // previous panel fully consumed
#pragma unroll
    for (int j = 0; j < 8; ++j) {             // publish staged panel to LDS
      int c = j * 128 + t;
      *(v8h*)(panel[0] + CHUNK_K(c) * PANEL_W + CHUNK_OFF(c)) = stg[j];
    }
    __syncthreads();                          // panel visible to all waves
    if (p + 1 < NPAN) LOAD_STAGE(C0 + (p + 1) * PANEL_W);
    const _Float16* cur = panel[0];
#endif

    v8f acc[2][2] = {};
#pragma unroll
    for (int kc = 0; kc < 8; ++kc) {
      v16h b0 = load_b_frag_lds(cur, kc * 32, 0);
      v16h b1 = load_b_frag_lds(cur, kc * 32, 16);
      acc[0][0] = wmma_f16(A[0][kc], b0, acc[0][0]);
      acc[1][0] = wmma_f16(A[1][kc], b0, acc[1][0]);   // reuse b0
      acc[0][1] = wmma_f16(A[0][kc], b1, acc[0][1]);
      acc[1][1] = wmma_f16(A[1][kc], b1, acc[1][1]);   // reuse b1
    }
#pragma unroll
    for (int mt = 0; mt < 2; ++mt)
#pragma unroll
      for (int nt = 0; nt < 2; ++nt)
#pragma unroll
        for (int r = 0; r < 8; ++r)
          sums[mt][r] += exp2f(acc[mt][nt][r] * SIM_SCALE);  // v_exp_f32

#if USE_ASYNC_LDS
    wait_asynccnt0();                         // next panel's DMA landed
    __syncthreads();                          // all waves done with cur + DMA done
#endif
  }

#pragma unroll
  for (int mt = 0; mt < 2; ++mt)
#pragma unroll
    for (int r = 0; r < 8; ++r)
      for (int off = 1; off < 16; off <<= 1)
        sums[mt][r] += __shfl_xor(sums[mt][r], off, 32);
  if ((lane & 15) == 0) {
    int mb = M0 + ((lane >> 4) << 3);
#pragma unroll
    for (int mt = 0; mt < 2; ++mt)
#pragma unroll
      for (int r = 0; r < 8; ++r)
        unsafeAtomicAdd(&S[mb + mt * 16 + r], sums[mt][r]);  // global_atomic_add_f32
  }
}

// ===========================================================================
// Kernel 4: diagonal dots + final loss.
// ===========================================================================
__global__ void finish_kernel(const _Float16* __restrict__ nh,
                              const float* __restrict__ S, float* __restrict__ out) {
  int i = blockIdx.x * blockDim.x + threadIdx.x;
  if (i >= NROW) return;
  const _Float16* p = nh + (size_t)i * DIM;
  const _Float16* q = nh + (size_t)(i + NROW) * DIM;
  float dot = 0.0f;
  for (int k = 0; k < DIM; ++k) dot += (float)p[k] * (float)q[k];
  float l1 = __logf(fmaxf(S[i] - E2, 1e-30f));
  float l2 = __logf(fmaxf(S[i + NROW] - E2, 1e-30f));
  out[i] = 0.5f * (l1 + l2) - dot * INV_TAU;
}

// ===========================================================================
extern "C" void kernel_launch(void* const* d_in, const int* in_sizes, int n_in,
                              void* d_out, int out_size, void* d_ws, size_t ws_size,
                              hipStream_t stream) {
  const float* z1 = (const float*)d_in[0];
  const float* z2 = (const float*)d_in[1];
  const float* W1 = (const float*)d_in[2];
  const float* b1 = (const float*)d_in[3];
  const float* W2 = (const float*)d_in[4];
  const float* b2 = (const float*)d_in[5];

  char* ws = (char*)d_ws;
  const size_t zbytes = (size_t)M2 * DIM * sizeof(_Float16);     // 8.4 MB
  _Float16* zh  = (_Float16*)ws;  ws += zbytes;
  _Float16* hh  = (_Float16*)ws;  ws += zbytes;
  _Float16* nh  = (_Float16*)ws;  ws += zbytes;
  _Float16* nT  = (_Float16*)ws;  ws += zbytes;
  _Float16* w1p = (_Float16*)ws;  ws += (size_t)DIM * DIM * sizeof(_Float16);
  _Float16* w2p = (_Float16*)ws;  ws += (size_t)DIM * DIM * sizeof(_Float16);
  float*    S   = (float*)ws;     ws += (size_t)M2 * sizeof(float);
  // total ~33.9 MB of d_ws

  prep_kernel<<<(2u * NROW * DIM) / 256, 256, 0, stream>>>(z1, z2, W1, W2,
                                                           zh, w1p, w2p, S);
  proj1_kernel<<<M2 / 128, 256, 0, stream>>>(zh, w1p, b1, hh);
  proj2_kernel<<<M2 / 64, 128, 0, stream>>>(hh, w2p, b2, nh, nT);
  sim_kernel<<<(M2 / 128) * 2, 128, 0, stream>>>(nh, nT, S);
  finish_kernel<<<NROW / 256, 256, 0, stream>>>(nh, S, (float*)d_out);
}
